// NonLocalAttention_20066087207409
// MI455X (gfx1250) — matrix operand order repaired and checked by reading, not compiled, not run on hardware
//
#include <hip/hip_runtime.h>
#include <hip/hip_bf16.h>
#include <cstdint>

// ---------------------------------------------------------------------------
// Types for CDNA5 WMMA (wave32): 16x16x32 bf16 -> f32
// ---------------------------------------------------------------------------
typedef __attribute__((ext_vector_type(16))) __bf16 v16bf;  // 8 VGPRs
typedef __attribute__((ext_vector_type(8)))  float  v8f;    // 8 VGPRs

union BFr { v16bf v; uint4 q[2]; };

__device__ __forceinline__ unsigned short f2bf(float f) {
    unsigned int u = __float_as_uint(f);
    u += 0x7FFFu + ((u >> 16) & 1u);           // round-to-nearest-even
    return (unsigned short)(u >> 16);
}
__device__ __forceinline__ float bf2f(unsigned short s) {
    return __uint_as_float(((unsigned int)s) << 16);
}

#define N_BATCH 4
#define C_IN    256
#define HW      4096

// --- CDNA5 async global->LDS path (ASYNCcnt), guarded so we always compile --
#if __has_builtin(__builtin_amdgcn_global_load_async_to_lds_b128) && \
    __has_builtin(__builtin_amdgcn_s_wait_asynccnt)
#define USE_ASYNC 1
// builtin expects: (int4 AS1* gaddr, int4 AS3* ldsaddr, imm offset, imm cpol)
typedef int v4i_t __attribute__((vector_size(16)));
typedef __attribute__((address_space(1))) v4i_t gv4i;
typedef __attribute__((address_space(3))) v4i_t lv4i;
// integer-routed addrspace casts (generic->global / generic->LDS-offset)
#define GPTR(p) ((gv4i*)(unsigned long long)(uintptr_t)(p))
#define LPTR(p) ((lv4i*)(unsigned int)(uintptr_t)(p))
#else
#define USE_ASYNC 0
#endif

// --- 16-lane XOR butterfly: v_permlane16_b32 (VALU) if available, else
// --- ds_bpermute via __shfl_xor.  Both halves of the wave use the same
// --- 16-lane permutation, which is exactly what the row-split C layout needs.
#if __has_builtin(__builtin_amdgcn_permlane16)
#define HAVE_PL16 1
#else
#define HAVE_PL16 0
#endif

__device__ __forceinline__ float xorswap16(float v, int step) {
#if HAVE_PL16
    unsigned s0, s1;
    switch (step) {                       // nibble i = i ^ step
        case 1:  s0 = 0x67452301u; s1 = 0xEFCDAB89u; break;
        case 2:  s0 = 0x54761032u; s1 = 0xDCFE98BAu; break;
        case 4:  s0 = 0x32107654u; s1 = 0xBA98FEDCu; break;
        default: s0 = 0xFEDCBA98u; s1 = 0x76543210u; break;
    }
    return __uint_as_float(__builtin_amdgcn_permlane16(
        __float_as_uint(v), __float_as_uint(v), s0, s1, false, false));
#else
    return __shfl_xor(v, step, 32);
#endif
}

// ---------------------------------------------------------------------------
// WMMA 1x1 conv + bias + PReLU, fp32 in, bf16 out.
//   Y[o,p] = prelu( sum_c W[o,c] * x[n,c,p] + b[o] )
// Block: 256 threads = 8 waves laid out 4 (pixel tiles) x 2 (output tiles);
// block tile = 64 pixels x 32 outputs, K = C_IN reduced in 32-chunks:
//   A = X^T (16p x 32c, bf16, transposed during LDS staging)
//   B = W^T (32c x 16o, bf16 rows straight from W[o][c])
// mode 0: dst[(n*HW + p)*ldm + off + o]           (pixel-major: Q / K layouts)
// mode 1: dst[(n*ldm + off + o)*HW + p]           (channel-major: V^T layout)
// ---------------------------------------------------------------------------
__global__ __launch_bounds__(256) void conv1x1_prelu_wmma_kernel(
    const float* __restrict__ x, const float* __restrict__ w,
    const float* __restrict__ bias, const float* __restrict__ alpha,
    unsigned short* __restrict__ dst,
    int mode, int ldm, int off, int accflag)
{
    __shared__ unsigned char lxb[64 * 80];   // X^T chunk [p][c] bf16, 80B rows
    __shared__ unsigned char lwb[32 * 80];   // W   chunk [o][c] bf16, 80B rows

    const int t  = threadIdx.x;
    const int wv = t >> 5;
    const int L  = t & 31;
    const int ln = L & 15;
    const int hi = L >> 4;
    const int wp = wv >> 1;          // 0..3: pixel tile of 16
    const int wo = wv & 1;           // 0..1: output tile of 16
    const int n  = blockIdx.z;
    const int p0 = blockIdx.x * 64;
    const int o0 = blockIdx.y * 32;

    const float* xb = x + (size_t)n * C_IN * HW;

    v8f acc = (v8f){0,0,0,0,0,0,0,0};

    for (int c0 = 0; c0 < C_IN; c0 += 32) {
        __syncthreads();
        // stage X^T: read x[c][p] as float4 (coalesced), scatter bf16 to [p][c]
#pragma unroll
        for (int i = 0; i < 2; i++) {
            int idx = t + 256 * i;           // 0..511 = 32c x 16 pixel-quads
            int c  = idx >> 4;
            int p4 = (idx & 15) * 4;
            float4 v = *(const float4*)(xb + (size_t)(c0 + c) * HW + p0 + p4);
            *(unsigned short*)(&lxb[(p4 + 0) * 80 + c * 2]) = f2bf(v.x);
            *(unsigned short*)(&lxb[(p4 + 1) * 80 + c * 2]) = f2bf(v.y);
            *(unsigned short*)(&lxb[(p4 + 2) * 80 + c * 2]) = f2bf(v.z);
            *(unsigned short*)(&lxb[(p4 + 3) * 80 + c * 2]) = f2bf(v.w);
        }
        // stage W: read w[o][c] as float4, store 4 packed bf16 (8B contiguous)
        {
            int o  = t >> 3;                 // 32 outputs
            int c4 = (t & 7) * 4;            // 32 channels / 4
            float4 v = *(const float4*)(w + (size_t)(o0 + o) * C_IN + c0 + c4);
            uint2 pk;
            pk.x = (unsigned)f2bf(v.x) | ((unsigned)f2bf(v.y) << 16);
            pk.y = (unsigned)f2bf(v.z) | ((unsigned)f2bf(v.w) << 16);
            *(uint2*)(&lwb[o * 80 + c4 * 2]) = pk;
        }
        __syncthreads();

        // A operand: X^T rows (A layout: lo lanes K{0..7,16..23}, hi {8..15,24..31})
        BFr a;
        const unsigned char* ab = &lxb[(wp * 16 + ln) * 80 + hi * 16];
        a.q[0] = *(const uint4*)ab;
        a.q[1] = *(const uint4*)(ab + 32);
        // B operand: W rows (B layout: lane = column o, lo lanes K{0..7,16..23}, hi {8..15,24..31})
        BFr b;
        const unsigned char* bb = &lwb[(wo * 16 + ln) * 80 + hi * 16];
        b.q[0] = *(const uint4*)bb;
        b.q[1] = *(const uint4*)(bb + 32);

        acc = __builtin_amdgcn_wmma_f32_16x16x32_bf16(
                  false, a.v, false, b.v, (short)0, acc, false, false);
    }

    // epilogue: bias + PReLU + (optional accumulate) + bf16 store
    const float al = alpha[0];
    const int o = o0 + wo * 16 + ln;
    const float bs = bias[o];
#pragma unroll
    for (int r = 0; r < 8; r++) {
        int p = p0 + wp * 16 + r + 8 * hi;
        float y = acc[r] + bs;
        y = (y >= 0.0f) ? y : al * y;
        size_t idx;
        if (mode == 0) idx = ((size_t)n * HW + p) * (size_t)ldm + off + o;
        else           idx = ((size_t)n * ldm + off + o) * (size_t)HW + p;
        if (accflag) y += bf2f(dst[idx]);
        dst[idx] = f2bf(y);
    }
}

// ---------------------------------------------------------------------------
// Flash attention (bf16 WMMA, f32 accumulate) with double-buffered LDS tiles
// fed by the CDNA5 async global->LDS engine.
//   Q  : bf16 [N][HW][64]   (r1 | t1)
//   Kc : bf16 [N][HW][64]   (t2 | r2)   -> S[q,k] = r1.t2 + t1.r2
//   Vs : bf16 [N][256][HW]  (V^T = prelu(ra)+prelu(ta), channel-major)
//   out: f32  [N][256][HW]
// 8 waves/block; each wave: 16 query rows x full 256-ch f32 accumulator.
// ---------------------------------------------------------------------------
#define KSZ   (64 * 144)            //  9216 B  K tile  [key][j]
#define VSZ   (256 * 144)           // 36864 B  V tile  [c][k]
#define BUFSZ (KSZ + VSZ)           // 46080 B  one K+V buffer
#define PSZ   (8 * 16 * 144)        // 18432 B  per-wave P scratch
#define SMEMSZ (2 * BUFSZ + PSZ)    // 110592 B dynamic LDS (<= 320KB/WGP)

// issue one 64-key tile copy: K (64x64 bf16) + V (256x64 bf16) -> LDS
__device__ __forceinline__ void issue_tile_copy(
    const uint4* __restrict__ gk, const uint4* __restrict__ gv,
    int n, int kt, char* kb, char* vb, int t)
{
    const int k0 = kt * 64;
#pragma unroll
    for (int i = 0; i < 2; i++) {
        int e = t + 256 * i;
        int r = e >> 3, s = e & 7;
        const uint4* src = gk + ((size_t)n * HW + k0 + r) * 8 + s;
        char* dst = kb + r * 144 + s * 16;
#if USE_ASYNC
        __builtin_amdgcn_global_load_async_to_lds_b128(GPTR(src), LPTR(dst), 0, 0);
#else
        *(uint4*)dst = *src;
#endif
    }
#pragma unroll
    for (int i = 0; i < 8; i++) {
        int e = t + 256 * i;
        int c = e >> 3, s = e & 7;
        const uint4* src = gv + ((size_t)n * 256 + c) * 512 + kt * 8 + s;
        char* dst = vb + c * 144 + s * 16;
#if USE_ASYNC
        __builtin_amdgcn_global_load_async_to_lds_b128(GPTR(src), LPTR(dst), 0, 0);
#else
        *(uint4*)dst = *src;
#endif
    }
}

__global__ __launch_bounds__(256, 1) void attn_kernel(
    const unsigned short* __restrict__ Q,
    const unsigned short* __restrict__ Kc,
    const unsigned short* __restrict__ Vs,
    float* __restrict__ out)
{
    extern __shared__ char smem[];

    const int t  = threadIdx.x;
    const int wv = t >> 5;
    const int L  = t & 31;
    const int ln = L & 15;
    const int hi = L >> 4;
    const int n  = blockIdx.z;
    const int q0 = (blockIdx.x * 8 + wv) * 16;

    const uint4* gk = (const uint4*)Kc;
    const uint4* gv = (const uint4*)Vs;
    char* myP = smem + 2 * BUFSZ + wv * (16 * 144);

    // ---- load Q A-operands (16x64 -> two 16x32 bf16 A tiles) once ----
    BFr qa[2];
    {
        const uint4* gq = (const uint4*)Q;               // 8 uint4 per row
        size_t base = ((size_t)n * HW + q0 + ln) * 8;
#pragma unroll
        for (int kc = 0; kc < 2; kc++) {
            // A layout: lanes 0-15 K={0..7,16..23}, lanes 16-31 K={8..15,24..31}
            qa[kc].q[0] = gq[base + kc * 4 + hi];
            qa[kc].q[1] = gq[base + kc * 4 + hi + 2];
        }
    }

    v8f oacc[16];
#pragma unroll
    for (int i = 0; i < 16; i++) oacc[i] = (v8f){0,0,0,0,0,0,0,0};
    float mrow[8], lrow[8];
#pragma unroll
    for (int r = 0; r < 8; r++) { mrow[r] = -1e30f; lrow[r] = 0.0f; }

    // prologue: kick off tile 0 into buffer 0
    issue_tile_copy(gk, gv, n, 0, smem, smem + KSZ, t);

#pragma unroll 1
    for (int kt = 0; kt < 64; kt++) {
        char* kb = smem + (kt & 1) * BUFSZ;
        char* vb = kb + KSZ;

        // tile kt is in flight from iteration kt-1 (or the prologue):
        // wait for this wave's async copies, then rendezvous with the block.
#if USE_ASYNC
        __builtin_amdgcn_s_wait_asynccnt(0);
#endif
        __syncthreads();

        // overlap: start fetching tile kt+1 into the other buffer
        if (kt < 63) {
            char* kb2 = smem + ((kt + 1) & 1) * BUFSZ;
            issue_tile_copy(gk, gv, n, kt + 1, kb2, kb2 + KSZ, t);
        }
        if (kt + 2 < 64) {   // L2 prefetch two tiles ahead (global_prefetch_b8)
            __builtin_prefetch(gk + ((size_t)n * HW + (kt + 2) * 64 + (t >> 2)) * 8, 0, 1);
            __builtin_prefetch(gv + ((size_t)n * 256 + t) * 512 + (kt + 2) * 8, 0, 1);
        }

        // ---- S(16x64) = Q @ K^T : 2 groups of (4 B-loads -> 4 WMMAs) ----
        const char* kbase = kb + ln * 144 + hi * 16;
        v8f sac[4];
#pragma unroll
        for (int h = 0; h < 2; h++) {
            BFr bk[4];
#pragma unroll
            for (int j = 0; j < 2; j++) {
#pragma unroll
                for (int kc = 0; kc < 2; kc++) {
                    const char* pb = kbase + (h * 2 + j) * 2304 + kc * 64;
                    bk[j * 2 + kc].q[0] = *(const uint4*)pb;
                    bk[j * 2 + kc].q[1] = *(const uint4*)(pb + 32);
                }
            }
#pragma unroll
            for (int j = 0; j < 2; j++) {
                v8f a = (v8f){0,0,0,0,0,0,0,0};
                a = __builtin_amdgcn_wmma_f32_16x16x32_bf16(
                        false, qa[0].v, false, bk[j * 2 + 0].v, (short)0, a, false, false);
                a = __builtin_amdgcn_wmma_f32_16x16x32_bf16(
                        false, qa[1].v, false, bk[j * 2 + 1].v, (short)0, a, false, false);
                sac[h * 2 + j] = a;
            }
        }

        // ---- online softmax over this key tile ----
        float corr[8];
#pragma unroll
        for (int r = 0; r < 8; r++) {
            float mx = fmaxf(fmaxf(sac[0][r], sac[1][r]), fmaxf(sac[2][r], sac[3][r]));
#pragma unroll
            for (int m = 1; m < 16; m <<= 1) mx = fmaxf(mx, xorswap16(mx, m));
            float mn = fmaxf(mrow[r], mx);
            corr[r]  = __expf(mrow[r] - mn);
            mrow[r]  = mn;
            float ps = 0.0f;
#pragma unroll
            for (int nt = 0; nt < 4; nt++) {
                float pv = __expf(sac[nt][r] - mn);
                ps += pv;
                // C layout (row = r + 8*hi, col = nt*16 + ln) -> row-major P
                *(unsigned short*)(myP + (r + 8 * hi) * 144 + (nt * 16 + ln) * 2) = f2bf(pv);
            }
#pragma unroll
            for (int m = 1; m < 16; m <<= 1) ps += xorswap16(ps, m);
            lrow[r] = lrow[r] * corr[r] + ps;
        }
        // rescale running output
#pragma unroll
        for (int ct = 0; ct < 16; ct++) {
#pragma unroll
            for (int r = 0; r < 8; r++) oacc[ct][r] *= corr[r];
        }

        // keep the wave-private P stores ordered before the reloads
        asm volatile("" ::: "memory");

        // ---- reload P as A-operands ----
        BFr pa[2];
#pragma unroll
        for (int kc = 0; kc < 2; kc++) {
            const char* pp = myP + ln * 144 + kc * 64 + hi * 16;
            pa[kc].q[0] = *(const uint4*)pp;
            pa[kc].q[1] = *(const uint4*)(pp + 32);
        }

        // ---- O(16x256) += P @ V : 8 groups of (4 B-loads -> 4 WMMAs) ----
        const char* vbase = vb + ln * 144 + hi * 16;
#pragma unroll
        for (int g = 0; g < 8; g++) {
            BFr bv[4];
#pragma unroll
            for (int i = 0; i < 2; i++) {
#pragma unroll
                for (int kc = 0; kc < 2; kc++) {
                    const char* pb = vbase + (g * 2 + i) * 2304 + kc * 64;
                    bv[i * 2 + kc].q[0] = *(const uint4*)pb;
                    bv[i * 2 + kc].q[1] = *(const uint4*)(pb + 32);
                }
            }
#pragma unroll
            for (int i = 0; i < 2; i++) {
                int ct = g * 2 + i;
                v8f a = oacc[ct];
                a = __builtin_amdgcn_wmma_f32_16x16x32_bf16(
                        false, pa[0].v, false, bv[i * 2 + 0].v, (short)0, a, false, false);
                a = __builtin_amdgcn_wmma_f32_16x16x32_bf16(
                        false, pa[1].v, false, bv[i * 2 + 1].v, (short)0, a, false, false);
                oacc[ct] = a;
            }
        }
    }

    // ---- normalize and store out[n][c][q] (f32, NCHW) ----
#pragma unroll
    for (int r = 0; r < 8; r++) {
        float inv = 1.0f / lrow[r];
        int q = q0 + r + 8 * hi;
#pragma unroll
        for (int ct = 0; ct < 16; ct++) {
            int c = ct * 16 + ln;
            out[((size_t)n * 256 + c) * HW + q] = oacc[ct][r] * inv;
        }
    }
}

// ---------------------------------------------------------------------------
// Host-side launch
// ---------------------------------------------------------------------------
extern "C" void kernel_launch(void* const* d_in, const int* in_sizes, int n_in,
                              void* d_out, int out_size, void* d_ws, size_t ws_size,
                              hipStream_t stream) {
    const float* rgb     = (const float*)d_in[0];
    const float* thermal = (const float*)d_in[1];
    const float* rw1 = (const float*)d_in[2];
    const float* rb1 = (const float*)d_in[3];
    const float* ra1 = (const float*)d_in[4];
    const float* rw2 = (const float*)d_in[5];
    const float* rb2 = (const float*)d_in[6];
    const float* ra2 = (const float*)d_in[7];
    const float* raw = (const float*)d_in[8];
    const float* rab = (const float*)d_in[9];
    const float* raa = (const float*)d_in[10];
    const float* tw1 = (const float*)d_in[11];
    const float* tb1 = (const float*)d_in[12];
    const float* ta1 = (const float*)d_in[13];
    const float* tw2 = (const float*)d_in[14];
    const float* tb2 = (const float*)d_in[15];
    const float* ta2 = (const float*)d_in[16];
    const float* taw = (const float*)d_in[17];
    const float* tab = (const float*)d_in[18];
    const float* taa = (const float*)d_in[19];

    // workspace: Q (2MB) | K (2MB) | Vsum (8MB), all bf16
    unsigned short* Q  = (unsigned short*)d_ws;
    unsigned short* Kc = Q  + (size_t)N_BATCH * HW * 64;
    unsigned short* Vs = Kc + (size_t)N_BATCH * HW * 64;

    dim3 blk(256);
    dim3 gsm(HW / 64, 1, N_BATCH);   // 32-output convs
    dim3 gbg(HW / 64, 8, N_BATCH);   // 256-output assembly convs

    // Q = [r1 | t1], K = [t2 | r2]  (pixel-major, ld = 64)
    conv1x1_prelu_wmma_kernel<<<gsm, blk, 0, stream>>>(rgb,     rw1, rb1, ra1, Q,  0, 64, 0,  0);
    conv1x1_prelu_wmma_kernel<<<gsm, blk, 0, stream>>>(thermal, tw1, tb1, ta1, Q,  0, 64, 32, 0);
    conv1x1_prelu_wmma_kernel<<<gsm, blk, 0, stream>>>(thermal, tw2, tb2, ta2, Kc, 0, 64, 0,  0);
    conv1x1_prelu_wmma_kernel<<<gsm, blk, 0, stream>>>(rgb,     rw2, rb2, ra2, Kc, 0, 64, 32, 0);
    // Vsum = prelu(ra) + prelu(ta)  (channel-major, ld = 256)
    conv1x1_prelu_wmma_kernel<<<gbg, blk, 0, stream>>>(rgb,     raw, rab, raa, Vs, 1, 256, 0, 0);
    conv1x1_prelu_wmma_kernel<<<gbg, blk, 0, stream>>>(thermal, taw, tab, taa, Vs, 1, 256, 0, 1);

    // flash attention: 32 blocks x 8 waves x 16 queries = 4096 queries/batch
    attn_kernel<<<dim3(HW / (16 * 8), 1, N_BATCH), blk, SMEMSZ, stream>>>(
        Q, Kc, Vs, (float*)d_out);
}
